// RENet_2284922601774
// MI455X (gfx1250) — matrix operand order repaired and checked by reading, not compiled, not run on hardware
//
#include <hip/hip_runtime.h>
#include <hip/hip_bf16.h>
#include <math.h>

typedef __attribute__((ext_vector_type(16))) _Float16 v16h;
typedef __attribute__((ext_vector_type(8)))  _Float16 h8;
typedef __attribute__((ext_vector_type(8)))  float    v8f;

#define WAY 5
#define NQ 150
#define CIN 640
#define HH 10
#define PIX 100          // H*H
#define FD 64            // feature dim after 1x1 conv
#define NPAIR (NQ * WAY) // 750
#define NBLK_SEP 240     // persistent blocks for sepconv layer1

__device__ __forceinline__ v8f wmma_f16(v16h a, v16h b, v8f c) {
  return __builtin_amdgcn_wmma_f32_16x16x32_f16(false, a, false, b, (short)0, c,
                                                false, false);
}

// A fragment (16x32 f16, row-major source, leading dim ldk; 16B-aligned rows).
// lane<16 : row=lane,    K = {k0..k0+15}
// lane>=16: row=lane-16, K = {k0+16..k0+31}
__device__ __forceinline__ v16h fragA_ld(const _Float16* base, int ldk, int k0,
                                         int rows_valid) {
  int lane = threadIdx.x & 31;
  int m = lane & 15, hi = lane >> 4;
  if (m < rows_valid) {
    const h8* p = (const h8*)(base + m * ldk + k0 + (hi ? 16 : 0));
    h8 lo = p[0];
    h8 up = p[1]; // +8 halfs
    return __builtin_shufflevector(lo, up, 0, 1, 2, 3, 4, 5, 6, 7, 8, 9, 10, 11,
                                   12, 13, 14, 15);
  }
  v16h z;
#pragma unroll
  for (int j = 0; j < 16; ++j) z[j] = (_Float16)0.f;
  return z;
}

// B fragment (32x16 f16 KxN); source has each column contiguous in K
// ([n][k], leading dim ldk; 32B-aligned columns).
// lane<16 : col=lane,    K = {k0..k0+15}
// lane>=16: col=lane-16, K = {k0+16..k0+31}
__device__ __forceinline__ v16h fragB_ld(const _Float16* base, int ldk, int k0,
                                         int cols_valid) {
  int lane = threadIdx.x & 31;
  int n = lane & 15, hi = lane >> 4;
  if (n < cols_valid)
    return *(const v16h*)(base + n * ldk + k0 + (hi ? 16 : 0));
  v16h z;
#pragma unroll
  for (int j = 0; j < 16; ++j) z[j] = (_Float16)0.f;
  return z;
}

// ---------------------------------------------------------------------------
// K1: mean-subtract + 1x1 conv (WMMA GEMM, K=640) + affine + relu + L2 norm.
// Outputs f16 features [pixel][64] and per-pixel channel mean.
// ---------------------------------------------------------------------------
__global__ void __launch_bounds__(128)
feat_kernel(const float* __restrict__ x, const float* __restrict__ wgt,
            const float* __restrict__ gam, const float* __restrict__ bet,
            int total_pix, _Float16* __restrict__ featOut,
            float* __restrict__ muOut) {
  __shared__ __align__(32) _Float16 aT[16 * 32];
  __shared__ __align__(32) _Float16 wT[64 * 32];
  __shared__ float red[16][8];
  __shared__ float mu[16];
  __shared__ float fbuf[16 * 64];
  __shared__ float inorm[16];

  const int tid = threadIdx.x;
  const int pix0 = blockIdx.x * 16;

  // thread -> (pixel i, 4 channels) mapping used for staging
  const int i = tid >> 3;
  const int c0 = (tid & 7) * 4;
  const int gp = pix0 + i;
  const bool pvalid = gp < total_pix;
  const int nimg = pvalid ? gp / PIX : 0;
  const int hw = pvalid ? gp - nimg * PIX : 0;
  const float* px = x + (size_t)nimg * CIN * PIX + hw; // safe base when invalid

  { // per-pixel channel mean over 640 channels
    float s = 0.f;
    if (pvalid) {
      int part = tid & 7;
      for (int c = part; c < CIN; c += 8) s += px[c * PIX];
    }
    red[i][tid & 7] = s;
  }
  __syncthreads();
  if (tid < 16) {
    float s = 0.f;
#pragma unroll
    for (int j = 0; j < 8; ++j) s += red[tid][j];
    float m = s * (1.f / (float)CIN);
    mu[tid] = m;
    int g2 = pix0 + tid;
    if (g2 < total_pix) muOut[g2] = m;
  }
  __syncthreads();

  const int wave = tid >> 5;
  const float* wrow = wgt + (size_t)(tid >> 1) * CIN + (tid & 1) * 16;
  _Float16* wdst = wT + (tid >> 1) * 32 + (tid & 1) * 16;

  v8f acc = {0.f, 0.f, 0.f, 0.f, 0.f, 0.f, 0.f, 0.f};

  for (int ks = 0; ks < CIN / 32; ++ks) {
    { // stage A: 16 pix x 32 ch, mean-subtracted (4 channels per thread)
      float mv = mu[i];
      float r0 = px[(ks * 32 + c0 + 0) * PIX];
      float r1 = px[(ks * 32 + c0 + 1) * PIX];
      float r2 = px[(ks * 32 + c0 + 2) * PIX];
      float r3 = px[(ks * 32 + c0 + 3) * PIX];
      aT[i * 32 + c0 + 0] = (_Float16)(pvalid ? r0 - mv : 0.f);
      aT[i * 32 + c0 + 1] = (_Float16)(pvalid ? r1 - mv : 0.f);
      aT[i * 32 + c0 + 2] = (_Float16)(pvalid ? r2 - mv : 0.f);
      aT[i * 32 + c0 + 3] = (_Float16)(pvalid ? r3 - mv : 0.f);
    }
#pragma unroll
    for (int t = 0; t < 16; ++t) // stage W: 64 out x 32 ch, 16 contiguous/thread
      wdst[t] = (_Float16)wrow[ks * 32 + t];
    __syncthreads();
    v16h a = fragA_ld(aT, 32, 0, 16);
    v16h b = fragB_ld(wT + wave * 16 * 32, 32, 0, 16);
    acc = wmma_f16(a, b, acc);
    __syncthreads();
  }

  { // affine + relu, transpose through LDS
    int lane = tid & 31;
    int hi = lane >> 4;
    int o = wave * 16 + (lane & 15);
#pragma unroll
    for (int r = 0; r < 8; ++r)
      fbuf[(r + hi * 8) * 64 + o] = fmaxf(gam[o] * acc[r] + bet[o], 0.f);
  }
  __syncthreads();
  if (tid < 16) {
    float ss = 0.f;
    for (int o = 0; o < 64; ++o) { float v = fbuf[tid * 64 + o]; ss += v * v; }
    inorm[tid] = 1.f / fmaxf(sqrtf(ss), 1e-8f);
  }
  __syncthreads();
#pragma unroll
  for (int t = 0; t < 8; ++t) {
    int e = tid + t * 128;
    int p = e >> 6, o = e & 63;
    int g2 = pix0 + p;
    if (g2 < total_pix)
      featOut[(size_t)g2 * FD + o] = (_Float16)(fbuf[e] * inorm[p]);
  }
}

// ---------------------------------------------------------------------------
// K2: per (q,w) pair: corr 100x100 = s (100x64) x q^T via WMMA into LDS,
// then SepConv4d layer0's two single-channel 3x3 convs -> z (f32, global).
// ---------------------------------------------------------------------------
__global__ void __launch_bounds__(256)
corr_sep0_kernel(const _Float16* __restrict__ sF, const _Float16* __restrict__ qF,
                 const float* __restrict__ w2c, const float* __restrict__ g2,
                 const float* __restrict__ b2, const float* __restrict__ w1c,
                 const float* __restrict__ g1, const float* __restrict__ b1,
                 float* __restrict__ zOut) {
  __shared__ float corr[PIX * PIX];    // 40 KB
  __shared__ _Float16 tbuf[PIX * PIX]; // 20 KB
  __shared__ float w2s[9], w1s[9], aff[4];

  const int b = blockIdx.x;
  const int qi = b / WAY, wi = b - qi * WAY;
  const int tid = threadIdx.x;
  const int wave = tid >> 5;
  const int lane = tid & 31;

  if (tid < 9) w2s[tid] = w2c[tid];
  else if (tid < 18) w1s[tid - 9] = w1c[tid - 9];
  else if (tid == 18) { aff[0] = g2[0]; aff[1] = b2[0]; aff[2] = g1[0]; aff[3] = b1[0]; }

  const _Float16* sBase = sF + (size_t)wi * PIX * FD;
  const _Float16* qBase = qF + (size_t)qi * PIX * FD;

  for (int p = wave; p < 49; p += 8) {
    int tm = p / 7, tn = p - tm * 7;
    int rv = min(16, PIX - tm * 16);
    int cv = min(16, PIX - tn * 16);
    v8f acc = {0.f, 0.f, 0.f, 0.f, 0.f, 0.f, 0.f, 0.f};
#pragma unroll
    for (int ks = 0; ks < 2; ++ks) {
      v16h a = fragA_ld(sBase + tm * 16 * FD, FD, ks * 32, rv);
      v16h bb = fragB_ld(qBase + tn * 16 * FD, FD, ks * 32, cv);
      acc = wmma_f16(a, bb, acc);
    }
    int hi = lane >> 4, nn = lane & 15;
    int col = tn * 16 + nn;
    if (col < PIX) {
#pragma unroll
      for (int r = 0; r < 8; ++r) {
        int row = tm * 16 + r + hi * 8;
        if (row < PIX) corr[row * PIX + col] = acc[r];
      }
    }
  }
  __syncthreads();

  // 3x3 conv over (u,v) (support spatial), affine(g2,b2), relu -> tbuf (f16)
  for (int idx = tid; idx < PIX * PIX; idx += 256) {
    int uv = idx / PIX, hw = idx - uv * PIX;
    int u = uv / HH, v = uv - u * HH;
    float s = 0.f;
#pragma unroll
    for (int du = -1; du <= 1; ++du)
#pragma unroll
      for (int dv = -1; dv <= 1; ++dv) {
        int uu = u + du, vv = v + dv;
        if ((unsigned)uu < HH && (unsigned)vv < HH)
          s += w2s[(du + 1) * 3 + (dv + 1)] * corr[(uu * HH + vv) * PIX + hw];
      }
    tbuf[idx] = (_Float16)fmaxf(aff[0] * s + aff[1], 0.f);
  }
  __syncthreads();

  // 3x3 conv over (h,w) (query spatial), affine(g1,b1) -> z (global)
  float* zb = zOut + (size_t)b * PIX * PIX;
  for (int idx = tid; idx < PIX * PIX; idx += 256) {
    int uv = idx / PIX, hw = idx - uv * PIX;
    int h = hw / HH, wq = hw - h * HH;
    float s = 0.f;
#pragma unroll
    for (int dh = -1; dh <= 1; ++dh)
#pragma unroll
      for (int dw = -1; dw <= 1; ++dw) {
        int h2 = h + dh, w2i = wq + dw;
        if ((unsigned)h2 < HH && (unsigned)w2i < HH)
          s += w1s[(dh + 1) * 3 + (dw + 1)] * (float)tbuf[uv * PIX + h2 * HH + w2i];
      }
    zb[idx] = aff[2] * s + aff[3];
  }
}

// ---------------------------------------------------------------------------
// K3: SepConv4d layer1 (dominant cost) as two implicit-im2col WMMA GEMMs.
// K is flattened TAP-MAJOR: k = tap*16 + ci, so each 16-wide B-fragment
// segment has ONE tap -> one bounds check, and for stage B the segment is one
// contiguous 32-byte row of t1[n][16ci] (global_load_b128 pair).
// x0[ci] = relu(a0[ci]*z + b0[ci])  (layer0 proj is rank-1 in channels).
// Stage A: conv over (u,v) -> t1[n][16] (f16 slice per block, L2 resident).
// Stage B: conv over (h,w) + folded affine/proj -> xOut (f32).
// ---------------------------------------------------------------------------
__global__ void __launch_bounds__(256)
sep1_kernel(const float* __restrict__ z, const float* __restrict__ s1w2,
            const float* __restrict__ s1g2, const float* __restrict__ s1b2,
            const float* __restrict__ s1w1, const float* __restrict__ s1g1,
            const float* __restrict__ s1b1, const float* __restrict__ s1wp,
            const float* __restrict__ s1gp, const float* __restrict__ s1bp,
            const float* __restrict__ s0wp, const float* __restrict__ s0gp,
            const float* __restrict__ s0bp, _Float16* __restrict__ t1ws,
            float* __restrict__ xOut) {
  __shared__ float zL[PIX * PIX];                 // 40 KB
  __shared__ __align__(32) _Float16 w2a[16 * 160]; // 5 KB (tap-major K)
  __shared__ __align__(32) _Float16 w1a[16 * 160]; // 5 KB (tap-major K)
  __shared__ float a0s[16], b0s[16], g2s[16], b2s[16], c1s[16];
  __shared__ float c0s;

  const int tid = threadIdx.x;
  const int wave = tid >> 5;
  const int lane = tid & 31;
  const int hi = lane >> 4;
  const int nloc = lane & 15;

  for (int e = tid; e < 16 * 160; e += 256) { // weights: [co][tap*16+ci]
    int co = e / 160, k = e - co * 160;
    int tap = k >> 4, ci = k & 15;
    float v2 = 0.f, v1 = 0.f;
    if (tap < 9) {
      v2 = s1w2[co * 144 + ci * 9 + tap];
      v1 = s1w1[co * 144 + ci * 9 + tap];
    }
    w2a[e] = (_Float16)v2;
    w1a[e] = (_Float16)v1;
  }
  if (tid < 16) {
    a0s[tid] = s0gp[tid] * s0wp[tid];
    b0s[tid] = s0bp[tid];
    g2s[tid] = s1g2[tid];
    b2s[tid] = s1b2[tid];
    c1s[tid] = s1gp[0] * s1wp[tid] * s1g1[tid];
  }
  if (tid == 0) {
    float s = 0.f;
    for (int co = 0; co < 16; ++co) s += s1wp[co] * s1b1[co];
    c0s = s1gp[0] * s + s1bp[0];
  }
  __syncthreads();

  // hoist all per-channel coefficients into registers
  float a0r[16], b0r[16];
#pragma unroll
  for (int j = 0; j < 16; ++j) { a0r[j] = a0s[j]; b0r[j] = b0s[j]; }
  float g2r[8], b2r[8], c1r[8];
#pragma unroll
  for (int r = 0; r < 8; ++r) {
    g2r[r] = g2s[r + hi * 8];
    b2r[r] = b2s[r + hi * 8];
    c1r[r] = c1s[r + hi * 8];
  }
  const float cc0 = c0s;

  v16h A2[5], A1[5]; // weight A-fragments in registers
#pragma unroll
  for (int ks = 0; ks < 5; ++ks) {
    A2[ks] = fragA_ld(w2a, 160, ks * 32, 16);
    A1[ks] = fragA_ld(w1a, 160, ks * 32, 16);
  }

  _Float16* t1 = t1ws + (size_t)blockIdx.x * 16 * PIX * PIX; // [n][16]

  for (int b = blockIdx.x; b < NPAIR; b += gridDim.x) {
    for (int e = tid; e < PIX * PIX; e += 256)
      zL[e] = z[(size_t)b * PIX * PIX + e];
    __syncthreads();

    // ---- Stage A: conv over (u,v) -------------------------------------
    for (int t = wave; t < 625; t += 8) {
      int n = t * 16 + nloc;
      int uv = n / PIX, hw = n - uv * PIX;
      int u = uv / HH, v = uv - u * HH;
      v8f acc = {0.f, 0.f, 0.f, 0.f, 0.f, 0.f, 0.f, 0.f};
#pragma unroll
      for (int ks = 0; ks < 5; ++ks) {
        int seg = ks * 2 + hi; // tap index; seg==9 is K padding
        bool ok = false;
        float zv = 0.f;
        if (seg < 9) {
          int uu = u + seg / 3 - 1;
          int vv = v + seg % 3 - 1;
          if ((unsigned)uu < HH && (unsigned)vv < HH) {
            ok = true;
            zv = zL[(uu * HH + vv) * PIX + hw];
          }
        }
        v16h bf;
#pragma unroll
        for (int j = 0; j < 16; ++j) {
          float val = ok ? fmaxf(a0r[j] * zv + b0r[j], 0.f) : 0.f;
          bf[j] = (_Float16)val;
        }
        acc = wmma_f16(A2[ks], bf, acc);
      }
      h8 st;
#pragma unroll
      for (int r = 0; r < 8; ++r)
        st[r] = (_Float16)fmaxf(g2r[r] * acc[r] + b2r[r], 0.f);
      *(h8*)(t1 + (size_t)n * 16 + hi * 8) = st; // packed 16B store
    }
    __threadfence();
    asm volatile("s_wait_storecnt 0x0" ::: "memory");
    __syncthreads();

    // ---- Stage B: conv over (h,w) + folded affine+proj ----------------
    for (int t = wave; t < 625; t += 8) {
      int n = t * 16 + nloc;
      int uv = n / PIX, hw = n - uv * PIX;
      int h = hw / HH, wq = hw - h * HH;
      v8f acc = {0.f, 0.f, 0.f, 0.f, 0.f, 0.f, 0.f, 0.f};
#pragma unroll
      for (int ks = 0; ks < 5; ++ks) {
        int seg = ks * 2 + hi;
        v16h bf;
        bool ok = false;
        int n2 = 0;
        if (seg < 9) {
          int h2 = h + seg / 3 - 1;
          int w2i = wq + seg % 3 - 1;
          if ((unsigned)h2 < HH && (unsigned)w2i < HH) {
            ok = true;
            n2 = uv * PIX + h2 * HH + w2i;
          }
        }
        if (ok) {
          bf = *(const v16h*)(t1 + (size_t)n2 * 16); // one 32B row load
        } else {
#pragma unroll
          for (int j = 0; j < 16; ++j) bf[j] = (_Float16)0.f;
        }
        acc = wmma_f16(A1[ks], bf, acc);
      }
      float part = 0.f;
#pragma unroll
      for (int r = 0; r < 8; ++r) part += c1r[r] * acc[r];
      part += __shfl_xor(part, 16, 32);
      if (hi == 0) xOut[(size_t)b * PIX * PIX + n] = part + cc0;
    }
    __syncthreads();
  }
}

// ---------------------------------------------------------------------------
// K4: gnorm (unbiased) + softmax attention over both axes, pooled cosine sim.
// ---------------------------------------------------------------------------
__global__ void __launch_bounds__(256)
attn_kernel(const float* __restrict__ xf, const float* __restrict__ spt,
            const float* __restrict__ qry, const float* __restrict__ muS,
            const float* __restrict__ muQ, float* __restrict__ out) {
  __shared__ float xL[PIX * PIX]; // 40 KB
  __shared__ float a_s[PIX], a_q[PIX];
  __shared__ float dm[2];
  __shared__ float rN[256], rA[256], rB[256];

  const int b = blockIdx.x;
  const int qi = b / WAY, wi = b - qi * WAY;
  const int tid = threadIdx.x;

  for (int e = tid; e < PIX * PIX; e += 256)
    xL[e] = xf[(size_t)b * PIX * PIX + e];
  if (tid < PIX) a_s[tid] = 0.f;
  else if (tid < 2 * PIX) a_q[tid - PIX] = 0.f;
  __syncthreads();

  if (tid < PIX) { // columns: normalize over uv, softmax over uv -> a_s
    int hw = tid;
    float s = 0.f, s2 = 0.f;
    for (int uv = 0; uv < PIX; ++uv) { float v = xL[uv * PIX + hw]; s += v; s2 += v * v; }
    float m = s * (1.f / PIX);
    float var = (s2 - (float)PIX * m * m) * (1.f / (PIX - 1));
    float inv = 1.f / (sqrtf(var + 1e-5f) * 5.0f); // fold 1/TEMP_ATTN
    float mx = -3.4e38f;
    for (int uv = 0; uv < PIX; ++uv) mx = fmaxf(mx, (xL[uv * PIX + hw] - m) * inv);
    float den = 0.f;
    for (int uv = 0; uv < PIX; ++uv) den += __expf((xL[uv * PIX + hw] - m) * inv - mx);
    float rden = 1.f / den;
    for (int uv = 0; uv < PIX; ++uv)
      atomicAdd(&a_s[uv], __expf((xL[uv * PIX + hw] - m) * inv - mx) * rden);
  } else if (tid < 2 * PIX) { // rows: normalize over hw, softmax over hw -> a_q
    const float* row = xL + (tid - PIX) * PIX;
    float s = 0.f, s2 = 0.f;
    for (int hw = 0; hw < PIX; ++hw) { float v = row[hw]; s += v; s2 += v * v; }
    float m = s * (1.f / PIX);
    float var = (s2 - (float)PIX * m * m) * (1.f / (PIX - 1));
    float inv = 1.f / (sqrtf(var + 1e-5f) * 5.0f);
    float mx = -3.4e38f;
    for (int hw = 0; hw < PIX; ++hw) mx = fmaxf(mx, (row[hw] - m) * inv);
    float den = 0.f;
    for (int hw = 0; hw < PIX; ++hw) den += __expf((row[hw] - m) * inv - mx);
    float rden = 1.f / den;
    for (int hw = 0; hw < PIX; ++hw)
      atomicAdd(&a_q[hw], __expf((row[hw] - m) * inv - mx) * rden);
  }
  __syncthreads();
  if (tid == 0) { float s = 0.f; for (int i = 0; i < PIX; ++i) s += a_s[i] * muS[wi * PIX + i]; dm[0] = s; }
  if (tid == 1) { float s = 0.f; for (int i = 0; i < PIX; ++i) s += a_q[i] * muQ[qi * PIX + i]; dm[1] = s; }
  __syncthreads();

  float num = 0.f, na = 0.f, nb = 0.f;
  for (int c = tid; c < CIN; c += 256) {
    const float* ps = spt + ((size_t)wi * CIN + c) * PIX;
    const float* pq = qry + ((size_t)qi * CIN + c) * PIX;
    float sa = 0.f, sb = 0.f;
    for (int i = 0; i < PIX; ++i) { sa += a_s[i] * ps[i]; sb += a_q[i] * pq[i]; }
    float spc = (sa - dm[0]) * (1.f / PIX);
    float qpc = (sb - dm[1]) * (1.f / PIX);
    num += spc * qpc; na += spc * spc; nb += qpc * qpc;
  }
  rN[tid] = num; rA[tid] = na; rB[tid] = nb;
  __syncthreads();
  for (int st = 128; st > 0; st >>= 1) {
    if (tid < st) { rN[tid] += rN[tid + st]; rA[tid] += rA[tid + st]; rB[tid] += rB[tid + st]; }
    __syncthreads();
  }
  if (tid == 0) {
    float d = fmaxf(sqrtf(rA[0]), 1e-8f) * fmaxf(sqrtf(rB[0]), 1e-8f);
    out[b] = (rN[0] / d) * 5.0f; // 1/TEMP
  }
}

// ---------------------------------------------------------------------------
extern "C" void kernel_launch(void* const* d_in, const int* in_sizes, int n_in,
                              void* d_out, int out_size, void* d_ws,
                              size_t ws_size, hipStream_t stream) {
  (void)in_sizes; (void)n_in; (void)out_size; (void)ws_size;
  const float* spt  = (const float*)d_in[0];
  const float* qry  = (const float*)d_in[1];
  const float* ccaw = (const float*)d_in[2];
  const float* ccag = (const float*)d_in[3];
  const float* ccab = (const float*)d_in[4];
  const float* s0w2 = (const float*)d_in[5];
  const float* s0g2 = (const float*)d_in[6];
  const float* s0b2 = (const float*)d_in[7];
  const float* s0w1 = (const float*)d_in[8];
  const float* s0g1 = (const float*)d_in[9];
  const float* s0b1 = (const float*)d_in[10];
  const float* s0wp = (const float*)d_in[11];
  const float* s0gp = (const float*)d_in[12];
  const float* s0bp = (const float*)d_in[13];
  const float* s1w2 = (const float*)d_in[14];
  const float* s1g2 = (const float*)d_in[15];
  const float* s1b2 = (const float*)d_in[16];
  const float* s1w1 = (const float*)d_in[17];
  const float* s1g1 = (const float*)d_in[18];
  const float* s1b1 = (const float*)d_in[19];
  const float* s1wp = (const float*)d_in[20];
  const float* s1gp = (const float*)d_in[21];
  const float* s1bp = (const float*)d_in[22];

  char* ws = (char*)d_ws;
  size_t off = 0;
  auto carve = [&](size_t bytes) -> void* {
    size_t o = (off + 255) & ~(size_t)255;
    off = o + bytes;
    return (void*)(ws + o);
  };
  _Float16* sF  = (_Float16*)carve((size_t)WAY * PIX * FD * sizeof(_Float16));
  _Float16* qF  = (_Float16*)carve((size_t)NQ * PIX * FD * sizeof(_Float16));
  float* muSp   = (float*)carve((size_t)WAY * PIX * sizeof(float));
  float* muQp   = (float*)carve((size_t)NQ * PIX * sizeof(float));
  float* zp     = (float*)carve((size_t)NPAIR * PIX * PIX * sizeof(float));
  _Float16* t1p = (_Float16*)carve((size_t)NBLK_SEP * 16 * PIX * PIX * sizeof(_Float16));
  float* xfp    = (float*)carve((size_t)NPAIR * PIX * PIX * sizeof(float));

  feat_kernel<<<(WAY * PIX + 15) / 16, 128, 0, stream>>>(
      spt, ccaw, ccag, ccab, WAY * PIX, sF, muSp);
  feat_kernel<<<(NQ * PIX + 15) / 16, 128, 0, stream>>>(
      qry, ccaw, ccag, ccab, NQ * PIX, qF, muQp);
  corr_sep0_kernel<<<NPAIR, 256, 0, stream>>>(
      sF, qF, s0w2, s0g2, s0b2, s0w1, s0g1, s0b1, zp);
  sep1_kernel<<<NBLK_SEP, 256, 0, stream>>>(
      zp, s1w2, s1g2, s1b2, s1w1, s1g1, s1b1, s1wp, s1gp, s1bp,
      s0wp, s0gp, s0bp, t1p, xfp);
  attn_kernel<<<NPAIR, 256, 0, stream>>>(
      xfp, spt, qry, muSp, muQp, (float*)d_out);
}
